// Seq2Seq_44246753083840
// MI455X (gfx1250) — matrix operand order repaired and checked by reading, not compiled, not run on hardware
//
#include <hip/hip_runtime.h>
#include <math.h>

// ---------------------------------------------------------------------------
// Problem constants (from the reference):
//   B=2048, T=100, IN=64, H=512, PRED=30, OUT=2, gates G3 = 3*H = 1536
// ---------------------------------------------------------------------------
#define BATCH    2048
#define TSTEPS   100
#define IN_DIM   64
#define HID      512
#define G3       1536
#define PRED_N   30
#define OUT_DIM  2
#define BT       16          // batch rows per workgroup (one WMMA M tile)
#define NWAVES   8           // 256 threads
#define COLS_PW  64          // hidden columns per wave (4 N tiles of 16)

typedef __attribute__((ext_vector_type(16))) __bf16 v16bf;
typedef __attribute__((ext_vector_type(8)))  float  v8f;

union BF16x16 { v16bf v; uint4 q[2]; };

// round-to-nearest-even fp32 -> bf16 (as raw u16)
static __device__ __forceinline__ unsigned short f2bf(float f) {
    unsigned u = __float_as_uint(f);
    u += 0x7fffu + ((u >> 16) & 1u);
    return (unsigned short)(u >> 16);
}

static __device__ __forceinline__ float sigmoidf_(float x) {
    return 1.0f / (1.0f + __expf(-x));
}

// ---------------------------------------------------------------------------
// A fragment (16x32 bf16, row-major source in LDS, leading dim ldk elements).
// ISA layout: lanes 0-15 -> M=lane, K blocks {0..7, 16..23}
//             lanes 16-31 -> M=lane-16, K blocks {8..15, 24..31}
// => per lane two contiguous 16B chunks at k0 and k0+16.
// ---------------------------------------------------------------------------
static __device__ __forceinline__ v16bf load_a(const unsigned short* A,
                                               int ldk, int kb, int lane) {
    const int m  = lane & 15;
    const int ks = (lane & 16) ? 8 : 0;
    const unsigned short* p = A + m * ldk + kb + ks;
    BF16x16 r;
    r.q[0] = *reinterpret_cast<const uint4*>(p);
    r.q[1] = *reinterpret_cast<const uint4*>(p + 16);
    return r.v;
}

// ---------------------------------------------------------------------------
// B fragment (32x16 bf16). Source: weight matrix W stored bf16 row-major
// [G3][ldk] (i.e. original torch [3H, K]); B column n = W row n.
// lanes 0-15: N=lane,    K = kb+{0..7, 16..23}
// lanes 16-31: N=lane-16, K = kb+{8..15, 24..31}
// => per lane two 16B chunks at k0 and k0+16.
// ---------------------------------------------------------------------------
static __device__ __forceinline__ v16bf load_b(const unsigned short* W,
                                               int ldk, int n, int kb, int lane) {
    const unsigned short* p = W + (size_t)n * ldk + kb + ((lane & 16) ? 8 : 0);
    BF16x16 r;
    r.q[0] = *reinterpret_cast<const uint4*>(p);
    r.q[1] = *reinterpret_cast<const uint4*>(p + 16);
    return r.v;
}

static __device__ __forceinline__ void zero4(v8f a[4]) {
#pragma unroll
    for (int ct = 0; ct < 4; ++ct)
#pragma unroll
        for (int i = 0; i < 8; ++i) a[ct][i] = 0.0f;
}

// ---------------------------------------------------------------------------
// Accumulate A(16xK) @ W^T into the three gate accumulators of this wave.
// ar += A@Wr^T, az += A@Wz^T, ax += A@Wn^T  (gate g occupies rows g*512.. of W)
// ---------------------------------------------------------------------------
static __device__ __forceinline__ void gemm3(const unsigned short* A, int K,
                                             const unsigned short* W,
                                             v8f ar[4], v8f az[4], v8f ax[4],
                                             int colbase, int lane) {
    for (int kb = 0; kb < K; kb += 32) {
        v16bf a = load_a(A, K, kb, lane);
#pragma unroll
        for (int ct = 0; ct < 4; ++ct) {
            const int n = colbase + ct * 16 + (lane & 15);
            v16bf b0 = load_b(W, K, n, kb, lane);
            ar[ct] = __builtin_amdgcn_wmma_f32_16x16x32_bf16(
                false, a, false, b0, (short)0, ar[ct], false, false);
            v16bf b1 = load_b(W, K, n + HID, kb, lane);
            az[ct] = __builtin_amdgcn_wmma_f32_16x16x32_bf16(
                false, a, false, b1, (short)0, az[ct], false, false);
            v16bf b2 = load_b(W, K, n + 2 * HID, kb, lane);
            ax[ct] = __builtin_amdgcn_wmma_f32_16x16x32_bf16(
                false, a, false, b2, (short)0, ax[ct], false, false);
        }
    }
}

// GRU gate math in the WMMA C layout; updates fp32 hidden state in registers.
static __device__ __forceinline__ void gru_update(v8f ar[4], v8f az[4],
                                                  v8f ain[4], v8f ahn[4],
                                                  const float* bih, const float* bhh,
                                                  v8f hs[4], int colbase, int lane) {
#pragma unroll
    for (int ct = 0; ct < 4; ++ct) {
        const int col = colbase + ct * 16 + (lane & 15);
        const float brb = bih[col] + bhh[col];
        const float bzb = bih[col + HID] + bhh[col + HID];
        const float bib = bih[col + 2 * HID];
        const float bhb = bhh[col + 2 * HID];
#pragma unroll
        for (int v = 0; v < 8; ++v) {
            const float r = sigmoidf_(ar[ct][v] + brb);
            const float z = sigmoidf_(az[ct][v] + bzb);
            const float n = tanhf(ain[ct][v] + bib + r * (ahn[ct][v] + bhb));
            hs[ct][v] = (1.0f - z) * n + z * hs[ct][v];
        }
    }
}

// Down-convert this wave's hidden columns to bf16 into LDS (row-major [BT][HID]).
static __device__ __forceinline__ void store_h(const v8f hs[4],
                                               unsigned short* h_lds,
                                               int colbase, int lane) {
    const int mbase = (lane & 16) ? 8 : 0;
#pragma unroll
    for (int ct = 0; ct < 4; ++ct) {
        const int col = colbase + ct * 16 + (lane & 15);
#pragma unroll
        for (int v = 0; v < 8; ++v)
            h_lds[(mbase + v) * HID + col] = f2bf(hs[ct][v]);
    }
}

// ---------------------------------------------------------------------------
// Weight convert/pad: fp32 [rows][sc] -> bf16 [rows][dc] (zero pad sc..dc)
// ---------------------------------------------------------------------------
__global__ void cvt_kernel(const float* __restrict__ src,
                           unsigned short* __restrict__ dst,
                           int rows, int sc, int dc) {
    int i = blockIdx.x * blockDim.x + threadIdx.x;
    int n = rows * dc;
    if (i >= n) return;
    int r = i / dc, c = i - r * dc;
    float v = (c < sc) ? src[(size_t)r * sc + c] : 0.0f;
    dst[i] = f2bf(v);
}

// ---------------------------------------------------------------------------
// Fused seq2seq: each workgroup owns 16 batch rows and runs encoder T=100
// steps (2 GRU layers) + decoder 30 steps, fully resident in LDS/regs.
// ---------------------------------------------------------------------------
__global__ __launch_bounds__(256) void seq2seq_kernel(
    const float* __restrict__ in_data,       // [B][T][IN]
    const float* __restrict__ last_location, // [B][1][OUT]
    const unsigned short* __restrict__ enc_wih0b,  // bf16 [G3][IN]
    const unsigned short* __restrict__ enc_whh0b,  // bf16 [G3][H]
    const unsigned short* __restrict__ enc_wih1b,  // bf16 [G3][H]
    const unsigned short* __restrict__ enc_whh1b,  // bf16 [G3][H]
    const unsigned short* __restrict__ dec_wih0b,  // bf16 [G3][32] (K padded 2->32)
    const unsigned short* __restrict__ dec_whh0b,  // bf16 [G3][H]
    const unsigned short* __restrict__ dec_wih1b,  // bf16 [G3][H]
    const unsigned short* __restrict__ dec_whh1b,  // bf16 [G3][H]
    const float* __restrict__ enc_bih0, const float* __restrict__ enc_bhh0,
    const float* __restrict__ enc_bih1, const float* __restrict__ enc_bhh1,
    const float* __restrict__ dec_bih0, const float* __restrict__ dec_bhh0,
    const float* __restrict__ dec_bih1, const float* __restrict__ dec_bhh1,
    const float* __restrict__ lin_w,  // [OUT][H]
    const float* __restrict__ lin_b,  // [OUT]
    float* __restrict__ out)          // [B][PRED][OUT]
{
    __shared__ __align__(16) unsigned short x_lds[BT * IN_DIM];  // 2 KB
    __shared__ __align__(16) unsigned short h0_lds[BT * HID];    // 16 KB
    __shared__ __align__(16) unsigned short h1_lds[BT * HID];    // 16 KB
    __shared__ __align__(16) unsigned short xd_lds[BT * 32];     // 1 KB
    __shared__ __align__(16) float th_lds[BT * HID];             // 32 KB
    __shared__ float red_lds[32 * 8];

    const int tid  = threadIdx.x;
    const int lane = tid & 31;
    const int wv   = tid >> 5;
    const int rowbase = blockIdx.x * BT;
    const int colbase = wv * COLS_PW;

    // zero hidden state (bf16 copies in LDS, fp32 in registers)
    for (int i = tid; i < BT * HID; i += 256) { h0_lds[i] = 0; h1_lds[i] = 0; }
    v8f h0s[4], h1s[4];
    zero4(h0s); zero4(h1s);

    v8f ar[4], az[4], ain[4], ahn[4];

    // ------------------------------ encoder ------------------------------
    for (int t = 0; t < TSTEPS; ++t) {
        // stage x_t (16x64 fp32 -> bf16 LDS); 256 threads x 4 elems
        {
            const int e = tid * 4;                 // < 1024
            const int m = e >> 6, i = e & 63;
            const float4 xf = *reinterpret_cast<const float4*>(
                in_data + (size_t)(rowbase + m) * TSTEPS * IN_DIM +
                (size_t)t * IN_DIM + i);
            uint2 pk;
            pk.x = (unsigned)f2bf(xf.x) | ((unsigned)f2bf(xf.y) << 16);
            pk.y = (unsigned)f2bf(xf.z) | ((unsigned)f2bf(xf.w) << 16);
            *reinterpret_cast<uint2*>(&x_lds[e]) = pk;
        }
        __syncthreads();

        // layer 0
        zero4(ar); zero4(az); zero4(ain); zero4(ahn);
        gemm3(x_lds, IN_DIM, enc_wih0b, ar, az, ain, colbase, lane);
        gemm3(h0_lds, HID,   enc_whh0b, ar, az, ahn, colbase, lane);
        gru_update(ar, az, ain, ahn, enc_bih0, enc_bhh0, h0s, colbase, lane);
        __syncthreads();                 // everyone finished reading h0_lds
        store_h(h0s, h0_lds, colbase, lane);
        __syncthreads();                 // new h0 visible

        // layer 1
        zero4(ar); zero4(az); zero4(ain); zero4(ahn);
        gemm3(h0_lds, HID, enc_wih1b, ar, az, ain, colbase, lane);
        gemm3(h1_lds, HID, enc_whh1b, ar, az, ahn, colbase, lane);
        gru_update(ar, az, ain, ahn, enc_bih1, enc_bhh1, h1s, colbase, lane);
        __syncthreads();                 // everyone finished reading h1_lds
        store_h(h1s, h1_lds, colbase, lane);
        __syncthreads();
    }

    // ------------------------------ decoder ------------------------------
    // x0 = last_location, zero-padded to K=32 bf16
    for (int i = tid; i < BT * 32; i += 256) {
        const int m = i >> 5, c = i & 31;
        const float v = (c < OUT_DIM)
            ? last_location[(size_t)(rowbase + m) * OUT_DIM + c] : 0.0f;
        xd_lds[i] = f2bf(v);
    }

    for (int p = 0; p < PRED_N; ++p) {
        __syncthreads();

        // layer 0
        zero4(ar); zero4(az); zero4(ain); zero4(ahn);
        gemm3(xd_lds, 32,  dec_wih0b, ar, az, ain, colbase, lane);
        gemm3(h0_lds, HID, dec_whh0b, ar, az, ahn, colbase, lane);
        gru_update(ar, az, ain, ahn, dec_bih0, dec_bhh0, h0s, colbase, lane);
        __syncthreads();
        store_h(h0s, h0_lds, colbase, lane);
        __syncthreads();

        // layer 1
        zero4(ar); zero4(az); zero4(ain); zero4(ahn);
        gemm3(h0_lds, HID, dec_wih1b, ar, az, ain, colbase, lane);
        gemm3(h1_lds, HID, dec_whh1b, ar, az, ahn, colbase, lane);
        gru_update(ar, az, ain, ahn, dec_bih1, dec_bhh1, h1s, colbase, lane);

        // tanh(h1) (fp32) for the output projection
        {
            const int mbase = (lane & 16) ? 8 : 0;
#pragma unroll
            for (int ct = 0; ct < 4; ++ct) {
                const int col = colbase + ct * 16 + (lane & 15);
#pragma unroll
                for (int v = 0; v < 8; ++v)
                    th_lds[(mbase + v) * HID + col] = tanhf(h1s[ct][v]);
            }
        }
        __syncthreads();                 // h1_lds reads done, th ready
        store_h(h1s, h1_lds, colbase, lane);

        // projection: out[m][j] = tanh(h1[m]) . lin_w[j] + lin_b[j]
        {
            const int mj = tid & 31;         // (m,j) pair
            const int m = mj & 15, j = mj >> 4;
            const int chunk = tid >> 5;      // 8 chunks of 64 columns
            float s = 0.0f;
            const float* wrow = lin_w + (size_t)j * HID + chunk * 64;
            const float* trow = th_lds + m * HID + chunk * 64;
#pragma unroll 8
            for (int c = 0; c < 64; ++c) s += trow[c] * wrow[c];
            red_lds[mj * 8 + chunk] = s;
        }
        __syncthreads();
        if (tid < 32) {
            const int m = tid & 15, j = tid >> 4;
            float s = lin_b[j];
#pragma unroll
            for (int k = 0; k < 8; ++k) s += red_lds[tid * 8 + k];
            out[((size_t)(rowbase + m) * PRED_N + p) * OUT_DIM + j] = s;
            xd_lds[m * 32 + j] = f2bf(s);    // feedback as next decoder input
        }
        // loop-top __syncthreads guards xd_lds / h1_lds for next step
    }
}

// ---------------------------------------------------------------------------
// Host launcher
// ---------------------------------------------------------------------------
extern "C" void kernel_launch(void* const* d_in, const int* in_sizes, int n_in,
                              void* d_out, int out_size, void* d_ws, size_t ws_size,
                              hipStream_t stream) {
    const float* in_data  = (const float*)d_in[0];
    const float* last_loc = (const float*)d_in[1];
    const float* enc_wih0 = (const float*)d_in[2];
    const float* enc_whh0 = (const float*)d_in[3];
    const float* enc_bih0 = (const float*)d_in[4];
    const float* enc_bhh0 = (const float*)d_in[5];
    const float* enc_wih1 = (const float*)d_in[6];
    const float* enc_whh1 = (const float*)d_in[7];
    const float* enc_bih1 = (const float*)d_in[8];
    const float* enc_bhh1 = (const float*)d_in[9];
    const float* dec_wih0 = (const float*)d_in[10];
    const float* dec_whh0 = (const float*)d_in[11];
    const float* dec_bih0 = (const float*)d_in[12];
    const float* dec_bhh0 = (const float*)d_in[13];
    const float* dec_wih1 = (const float*)d_in[14];
    const float* dec_whh1 = (const float*)d_in[15];
    const float* dec_bih1 = (const float*)d_in[16];
    const float* dec_bhh1 = (const float*)d_in[17];
    const float* lin_w    = (const float*)d_in[18];
    const float* lin_b    = (const float*)d_in[19];
    float* out = (float*)d_out;

    // bf16 weight arena in d_ws
    unsigned short* ws = (unsigned short*)d_ws;
    unsigned short* w_eih0 = ws;                         // [G3][64]
    unsigned short* w_ehh0 = w_eih0 + (size_t)G3 * IN_DIM;
    unsigned short* w_eih1 = w_ehh0 + (size_t)G3 * HID;
    unsigned short* w_ehh1 = w_eih1 + (size_t)G3 * HID;
    unsigned short* w_dih0 = w_ehh1 + (size_t)G3 * HID;  // [G3][32] padded
    unsigned short* w_dhh0 = w_dih0 + (size_t)G3 * 32;
    unsigned short* w_dih1 = w_dhh0 + (size_t)G3 * HID;
    unsigned short* w_dhh1 = w_dih1 + (size_t)G3 * HID;

    auto cvt = [&](const float* s, unsigned short* d, int rows, int sc, int dc) {
        int n = rows * dc;
        cvt_kernel<<<(n + 255) / 256, 256, 0, stream>>>(s, d, rows, sc, dc);
    };
    cvt(enc_wih0, w_eih0, G3, IN_DIM, IN_DIM);
    cvt(enc_whh0, w_ehh0, G3, HID, HID);
    cvt(enc_wih1, w_eih1, G3, HID, HID);
    cvt(enc_whh1, w_ehh1, G3, HID, HID);
    cvt(dec_wih0, w_dih0, G3, OUT_DIM, 32);
    cvt(dec_whh0, w_dhh0, G3, HID, HID);
    cvt(dec_wih1, w_dih1, G3, HID, HID);
    cvt(dec_whh1, w_dhh1, G3, HID, HID);

    seq2seq_kernel<<<BATCH / BT, 256, 0, stream>>>(
        in_data, last_loc,
        w_eih0, w_ehh0, w_eih1, w_ehh1,
        w_dih0, w_dhh0, w_dih1, w_dhh1,
        enc_bih0, enc_bhh0, enc_bih1, enc_bhh1,
        dec_bih0, dec_bhh0, dec_bih1, dec_bhh1,
        lin_w, lin_b, out);
}
